// RecurrentLinearMHA_17514876634073
// MI455X (gfx1250) — compile-verified, matrix operand order checked by reading, not run on hardware
//
#include <hip/hip_runtime.h>
#include <cstdint>
#include <cstddef>

// ---------------------------------------------------------------------------
// Types
// ---------------------------------------------------------------------------
typedef unsigned short u16;
typedef __attribute__((ext_vector_type(16))) __bf16 v16bf;
typedef __attribute__((ext_vector_type(8)))  float  v8f;

struct alignas(16) U4 { uint32_t x, y, z, w; };   // POD 16-byte vector load/store

#define EPSV 1e-6f

__device__ __forceinline__ u16 f2bf(float f) {
  union { float f; uint32_t u; } x; x.f = f;
  uint32_t u = x.u;
  u += 0x7FFFu + ((u >> 16) & 1u);   // round-to-nearest-even
  return (u16)(u >> 16);
}
__device__ __forceinline__ float bf2f(u16 h) {
  union { uint32_t u; float f; } x; x.u = ((uint32_t)h) << 16;
  return x.f;
}

union FragU { v16bf v; U4 q[2]; };

// 16x32 bf16 operand fragment, "A layout": lane L -> row = L&15,
// K base = ((L>>4)<<3); elems 0-7 = K..K+7, elems 8-15 = K+16..K+23.
// Used for the B operand too by reading the *transposed* matrix
// (BT row-major, row = column of B), which has the identical pattern.
__device__ __forceinline__ v16bf load_frag(const u16* base, int row_stride) {
  const int lane = threadIdx.x & 31;
  const u16* p = base + (lane & 15) * row_stride + ((lane >> 4) << 3);
  FragU f;
  f.q[0] = *(const U4*)(p);
  f.q[1] = *(const U4*)(p + 16);
  return f.v;
}

__device__ __forceinline__ v8f wmma_bf16(v16bf a, v16bf b, v8f c) {
  return __builtin_amdgcn_wmma_f32_16x16x32_bf16(false, a, false, b,
                                                 (short)0, c, false, false);
}

// ---------------------------------------------------------------------------
// Tensor Data Mover: 2D bf16 tile DMA global -> LDS (CDNA5 TDM, §8 of ISA).
// Descriptor built per CDNA5 ISA.  This toolchain exposes
// the 6-arg builtin: (u32x4 g0, i32x8 g1, i32x4 g2, i32x4 g3, i32x8, i32 cpol).
// Issue from ONE wave only (EXEC ignored by TDM); complete with
// s_wait_tensorcnt + workgroup barrier.
// ---------------------------------------------------------------------------
#if __has_builtin(__builtin_amdgcn_tensor_load_to_lds) && \
    __has_builtin(__builtin_amdgcn_s_wait_tensorcnt)
#define HAVE_TDM 1
#else
#define HAVE_TDM 0
#endif

#if HAVE_TDM
typedef __attribute__((ext_vector_type(4))) unsigned int tdm_v4u;
typedef __attribute__((ext_vector_type(8))) int          tdm_v8i;
typedef __attribute__((ext_vector_type(4))) int          tdm_v4i;

__device__ __forceinline__ void tdm_load_2d(const u16* gptr, u16* lptr,
                                            uint32_t rows, uint32_t cols,
                                            uint32_t row_stride_elems) {
  const uint64_t ga = (uint64_t)(uintptr_t)gptr;        // flat == global VA
  const uint32_t la = (uint32_t)(uintptr_t)lptr;        // low 32 bits == LDS offset
  const uint32_t td0 = row_stride_elems;                // tensor dim0 (elems)
  const uint32_t td1 = 1u << 20;                        // plenty of rows (no clip)

  // D# group 0: count=1 | lds_addr | global_addr[56:0] | type=2 ("image")
  tdm_v4u g0;
  g0[0] = 1u;                                           // count=1, user desc
  g0[1] = la;
  g0[2] = (uint32_t)(ga & 0xFFFFFFFFu);
  g0[3] = (uint32_t)((ga >> 32) & 0x01FFFFFFu) | 0x80000000u;

  // D# group 1: mask=0, data_size=2B, dims/strides in data_size units
  tdm_v8i g1;
  g1[0] = 0x00010000;                                   // data_size = 1 (2 bytes)
  g1[1] = (int)((td0 & 0xFFFFu) << 16);                 // tensor_dim0[15:0]
  g1[2] = (int)(((td0 >> 16) & 0xFFFFu) | ((td1 & 0xFFFFu) << 16));
  g1[3] = (int)(((td1 >> 16) & 0xFFFFu) | (cols << 16));// tile_dim0 = cols
  g1[4] = (int)(rows & 0xFFFFu);                        // tile_dim1 = rows, tile_dim2=0
  g1[5] = (int)row_stride_elems;                        // tensor_dim0_stride lo32
  g1[6] = 0;                                            // stride hi / dim1_stride lo
  g1[7] = 0;

  // D# groups 2/3: benign 2D settings (tensor_dim2=1, tensor_dim4=1, rest 0)
  tdm_v4i g2; g2[0] = 1; g2[1] = 0; g2[2] = 0; g2[3] = 0;
  tdm_v4i g3; g3[0] = 0; g3[1] = 0x00010000; g3[2] = 0; g3[3] = 0;
  tdm_v8i gz; gz[0] = 0; gz[1] = 0; gz[2] = 0; gz[3] = 0;
             gz[4] = 0; gz[5] = 0; gz[6] = 0; gz[7] = 0;

  __builtin_amdgcn_tensor_load_to_lds(g0, g1, g2, g3, gz, 0);
}
#endif

// ---------------------------------------------------------------------------
// f32 -> bf16 elementwise convert
// ---------------------------------------------------------------------------
__global__ void cvt_x_kernel(const float* __restrict__ in, u16* __restrict__ out,
                             size_t n) {
  size_t i = (size_t)blockIdx.x * blockDim.x + threadIdx.x;
  if (i < n) out[i] = f2bf(in[i]);
}

// f32 [D x D] -> bf16 transpose: WT[n][k] = W[k][n]; LDS 32x32 tiles.
__global__ void cvt_wT_kernel(const float* __restrict__ W, u16* __restrict__ WT,
                              int Dm) {
  __shared__ float t[32][33];
  const int bx = blockIdx.x * 32;   // n tile
  const int by = blockIdx.y * 32;   // k tile
  const int x = threadIdx.x;        // 0..31
  for (int y = threadIdx.y; y < 32; y += 8)
    t[y][x] = W[(size_t)(by + y) * Dm + bx + x];      // t[k_local][n_local]
  __syncthreads();
  for (int y = threadIdx.y; y < 32; y += 8)
    WT[(size_t)(bx + y) * Dm + by + x] = f2bf(t[x][y]);
}

// ---------------------------------------------------------------------------
// bf16 GEMM: C[M,N] = A[M,K] @ B[K,N], BT stored row-major [N,K].
// mode 0: bf16 out.  mode 1: phi()=elu+1 then bf16 out.  mode 2: f32 out.
// 256 threads = 8 waves; 128x128 workgroup tile; wave w -> rows 16w..16w+15.
// Double-buffered LDS: wave 0 issues TDM DMA for step+1, all waves compute
// WMMAs on the current buffer, then s_wait_tensorcnt + barrier.
// ---------------------------------------------------------------------------
#define GT_M 128
#define GT_N 128
#define GT_K 32

__global__ __launch_bounds__(256) void gemm_bf16_kernel(
    const u16* __restrict__ A, const u16* __restrict__ BT,
    u16* __restrict__ Cb, float* __restrict__ Cf,
    int M, int N, int K, int mode) {
  __shared__ __align__(16) u16 As[2][GT_M][GT_K];
  __shared__ __align__(16) u16 Bs[2][GT_N][GT_K];

  const int wg_m = blockIdx.x * GT_M;
  const int wg_n = blockIdx.y * GT_N;
  const int tid  = threadIdx.x;
  const int wave = tid >> 5;
  const int lane = tid & 31;
  const int nk   = K / GT_K;

  v8f acc[8];
#pragma unroll
  for (int i = 0; i < 8; ++i) acc[i] = (v8f){};

#if HAVE_TDM
  if (wave == 0) {
    tdm_load_2d(&A[(size_t)wg_m * K],  &As[0][0][0], GT_M, GT_K, (uint32_t)K);
    tdm_load_2d(&BT[(size_t)wg_n * K], &Bs[0][0][0], GT_N, GT_K, (uint32_t)K);
    __builtin_amdgcn_s_wait_tensorcnt(0);
  }
  __syncthreads();
#endif

  for (int step = 0; step < nk; ++step) {
    const int cur = step & 1;
    const int k0  = step * GT_K;

#if HAVE_TDM
    if (wave == 0 && step + 1 < nk) {   // kick DMA for next k-step (async)
      tdm_load_2d(&A[(size_t)wg_m * K + k0 + GT_K],  &As[cur ^ 1][0][0],
                  GT_M, GT_K, (uint32_t)K);
      tdm_load_2d(&BT[(size_t)wg_n * K + k0 + GT_K], &Bs[cur ^ 1][0][0],
                  GT_N, GT_K, (uint32_t)K);
    }
#else
    // synchronous fallback staging (2 x 16B per thread per tile)
    {
      const int i0 = tid, i1 = tid + 256;
      const int r0 = i0 / 4, c0 = (i0 % 4) * 8;
      const int r1 = i1 / 4, c1 = (i1 % 4) * 8;
      *(U4*)&As[cur][r0][c0] = *(const U4*)&A[(size_t)(wg_m + r0) * K + k0 + c0];
      *(U4*)&As[cur][r1][c1] = *(const U4*)&A[(size_t)(wg_m + r1) * K + k0 + c1];
      *(U4*)&Bs[cur][r0][c0] = *(const U4*)&BT[(size_t)(wg_n + r0) * K + k0 + c0];
      *(U4*)&Bs[cur][r1][c1] = *(const U4*)&BT[(size_t)(wg_n + r1) * K + k0 + c1];
    }
    __syncthreads();
#endif

    const v16bf afrag = load_frag(&As[cur][wave * 16][0], GT_K);
#pragma unroll
    for (int nt = 0; nt < 8; ++nt) {
      const v16bf bfrag = load_frag(&Bs[cur][nt * 16][0], GT_K);
      acc[nt] = wmma_bf16(afrag, bfrag, acc[nt]);
    }

#if HAVE_TDM
    if (wave == 0 && step + 1 < nk) __builtin_amdgcn_s_wait_tensorcnt(0);
#endif
    __syncthreads();
  }

  // Epilogue: lane L holds rows m = wave*16 + (L>>4)*8 + r, col n = nt*16 + (L&15)
  const int cl = lane & 15;
  const int rb = (lane >> 4) * 8;
#pragma unroll
  for (int nt = 0; nt < 8; ++nt) {
    const int n = wg_n + nt * 16 + cl;
#pragma unroll
    for (int r = 0; r < 8; ++r) {
      const int m = wg_m + wave * 16 + rb + r;
      float v = acc[nt][r];
      if (mode == 1) v = (v > 0.f) ? (v + 1.f) : __expf(v);  // elu(v)+1
      if (mode == 2) Cf[(size_t)m * N + n] = v;
      else           Cb[(size_t)m * N + n] = f2bf(v);
    }
  }
}

// ---------------------------------------------------------------------------
// Chunked causal linear attention. One workgroup per (b,h); 128 thr = 4 waves.
// State S [dk=64 x dv=64] f32 in LDS (+ bf16 transposed copy for WMMA B-use),
// normalizer z[64] f32.  32 chunks of 64 tokens, each chunk = pure WMMA math.
// Q/K/V chunk tiles are DMA'd by the TDM, double-buffered across chunks.
// ---------------------------------------------------------------------------
__global__ __launch_bounds__(128) void attn_chunk_kernel(
    const u16* __restrict__ Q, const u16* __restrict__ Km,
    const u16* __restrict__ V, u16* __restrict__ O,
    int S, int Dm, int H) {
  const int bh = blockIdx.x;
  const int b = bh / H, h = bh % H;
  const size_t rowbase = (size_t)b * S;
  const int colbase = h * 64;

  __shared__ __align__(16) u16 Qs[2][64][64];  // Q chunk   [token][dk]
  __shared__ __align__(16) u16 Ks[2][64][64];  // K chunk   [token][dk]
  __shared__ __align__(16) u16 Vs[2][64][64];  // V chunk   [token][dv]
  __shared__ __align__(16) u16 KTs[64][64];    // K^T       [dk][token]
  __shared__ __align__(16) u16 VTs[64][64];    // V^T       [dv][token]
  __shared__ __align__(16) u16 Ps[64][64];     // masked probs, bf16
  __shared__ __align__(16) u16 STs[64][64];    // S^T bf16  [dv][dk]
  __shared__ float Sf[64][64];                 // state f32 [dk][dv]
  __shared__ float zf[64];
  __shared__ float den[64];

  const int tid = threadIdx.x;
  const int wave = tid >> 5;
  const int lane = tid & 31;
  const int cl = lane & 15;
  const int rb = (lane >> 4) * 8;

  for (int i = tid; i < 64 * 64; i += 128) { (&Sf[0][0])[i] = 0.f; (&STs[0][0])[i] = 0; }
  if (tid < 64) zf[tid] = 0.f;

  const int nchunks = S / 64;

#if HAVE_TDM
  if (wave == 0) {   // prologue: DMA chunk 0 into buffer 0
    const size_t g0 = rowbase * Dm + colbase;
    tdm_load_2d(&Q[g0],  &Qs[0][0][0], 64, 64, (uint32_t)Dm);
    tdm_load_2d(&Km[g0], &Ks[0][0][0], 64, 64, (uint32_t)Dm);
    tdm_load_2d(&V[g0],  &Vs[0][0][0], 64, 64, (uint32_t)Dm);
    __builtin_amdgcn_s_wait_tensorcnt(0);
  }
#endif
  __syncthreads();

  for (int c = 0; c < nchunks; ++c) {
    const int cur = c & 1;
    const size_t r0 = rowbase + (size_t)c * 64;

#if HAVE_TDM
    if (wave == 0 && c + 1 < nchunks) {  // kick DMA for next chunk (async)
      const size_t gn = (r0 + 64) * Dm + colbase;
      tdm_load_2d(&Q[gn],  &Qs[cur ^ 1][0][0], 64, 64, (uint32_t)Dm);
      tdm_load_2d(&Km[gn], &Ks[cur ^ 1][0][0], 64, 64, (uint32_t)Dm);
      tdm_load_2d(&V[gn],  &Vs[cur ^ 1][0][0], 64, 64, (uint32_t)Dm);
    }
#else
    for (int i = tid; i < (64 * 64) / 8; i += 128) {
      const int row = i / 8, col = (i % 8) * 8;
      const size_t g = (r0 + row) * Dm + colbase + col;
      *(U4*)&Qs[cur][row][col] = *(const U4*)&Q[g];
      *(U4*)&Ks[cur][row][col] = *(const U4*)&Km[g];
      *(U4*)&Vs[cur][row][col] = *(const U4*)&V[g];
    }
    __syncthreads();
#endif

    // ---- build K^T / V^T in LDS ----
    for (int i = tid; i < 64 * 64; i += 128) {
      const int row = i >> 6, col = i & 63;
      KTs[col][row] = Ks[cur][row][col];
      VTs[col][row] = Vs[cur][row][col];
    }
    __syncthreads();

    // ---- scores = Q @ K^T, causal mask (inclusive), write P bf16 ----
    {
      v8f sc[4];
#pragma unroll
      for (int nt = 0; nt < 4; ++nt) {
        v8f a = (v8f){};
#pragma unroll
        for (int kk2 = 0; kk2 < 64; kk2 += 32)
          a = wmma_bf16(load_frag(&Qs[cur][wave * 16][kk2], 64),
                        load_frag(&Ks[cur][nt * 16][kk2], 64), a);
        sc[nt] = a;
      }
#pragma unroll
      for (int nt = 0; nt < 4; ++nt) {
        const int n = nt * 16 + cl;
#pragma unroll
        for (int r = 0; r < 8; ++r) {
          const int m = wave * 16 + rb + r;
          float v = sc[nt][r];
          if (n > m) v = 0.f;            // tau <= t (state updated before out)
          Ps[m][n] = f2bf(v);
        }
      }
    }
    __syncthreads();

    // ---- den[t] = rowsum(P) + q_t . z_prev + eps ----
    if (tid < 64) {
      float s = 0.f, qz = 0.f;
#pragma unroll 8
      for (int j = 0; j < 64; ++j) {
        s  += bf2f(Ps[tid][j]);
        qz += bf2f(Qs[cur][tid][j]) * zf[j];
      }
      den[tid] = s + qz + EPSV;
    }
    __syncthreads();

    // ---- out = (P @ V + Q @ S_prev) / den ----
    {
      v8f oacc[4];
#pragma unroll
      for (int nt = 0; nt < 4; ++nt) {
        v8f a = (v8f){};
#pragma unroll
        for (int kk2 = 0; kk2 < 64; kk2 += 32)
          a = wmma_bf16(load_frag(&Ps[wave * 16][kk2], 64),
                        load_frag(&VTs[nt * 16][kk2], 64), a);
#pragma unroll
        for (int kk2 = 0; kk2 < 64; kk2 += 32)
          a = wmma_bf16(load_frag(&Qs[cur][wave * 16][kk2], 64),
                        load_frag(&STs[nt * 16][kk2], 64), a);
        oacc[nt] = a;
      }
#pragma unroll
      for (int nt = 0; nt < 4; ++nt) {
        const int n = nt * 16 + cl;
#pragma unroll
        for (int r = 0; r < 8; ++r) {
          const int m = wave * 16 + rb + r;
          O[(r0 + m) * Dm + colbase + n] = f2bf(oacc[nt][r] / den[m]);
        }
      }
    }
    __syncthreads();   // all reads of STs / Ps done before state update

    // ---- state update: S += K^T @ V ; z += colsum(K) ----
    {
      v8f dacc[4];
#pragma unroll
      for (int nt = 0; nt < 4; ++nt) {
        v8f a = (v8f){};
#pragma unroll
        for (int kk2 = 0; kk2 < 64; kk2 += 32)
          a = wmma_bf16(load_frag(&KTs[wave * 16][kk2], 64),
                        load_frag(&VTs[nt * 16][kk2], 64), a);
        dacc[nt] = a;
      }
      if (tid < 64) {
        float s = zf[tid];
#pragma unroll 8
        for (int j = 0; j < 64; ++j) s += bf2f(KTs[tid][j]);
        zf[tid] = s;
      }
#pragma unroll
      for (int nt = 0; nt < 4; ++nt) {
        const int n = nt * 16 + cl;                    // dv
#pragma unroll
        for (int r = 0; r < 8; ++r) {
          const int m = wave * 16 + rb + r;            // dk
          const float ns = Sf[m][n] + dacc[nt][r];
          Sf[m][n] = ns;
          STs[n][m] = f2bf(ns);
        }
      }
    }

#if HAVE_TDM
    if (wave == 0 && c + 1 < nchunks) __builtin_amdgcn_s_wait_tensorcnt(0);
#endif
    __syncthreads();   // state/z visible; next chunk's tiles ready
  }
}

// ---------------------------------------------------------------------------
// Host-side launch
// ---------------------------------------------------------------------------
extern "C" void kernel_launch(void* const* d_in, const int* in_sizes, int n_in,
                              void* d_out, int out_size, void* d_ws, size_t ws_size,
                              hipStream_t stream) {
  (void)in_sizes; (void)n_in; (void)out_size; (void)ws_size;
  const int Bb = 4, S = 2048, Dm = 1024, H = 16;
  const size_t MS = (size_t)Bb * S;            // 8192 rows
  const size_t ND = MS * Dm;                   // 8,388,608 elems
  const size_t WD = (size_t)Dm * Dm;           // 1,048,576 elems

  const float* x  = (const float*)d_in[0];
  const float* Wq = (const float*)d_in[1];
  const float* Wk = (const float*)d_in[2];
  const float* Wv = (const float*)d_in[3];
  const float* Wo = (const float*)d_in[4];
  float* out = (float*)d_out;

  u16* ws  = (u16*)d_ws;
  u16* xb  = ws;              // x bf16                [MS x D]
  u16* WqT = xb  + ND;        // Wq^T bf16             [D x D]
  u16* WkT = WqT + WD;
  u16* WvT = WkT + WD;
  u16* WoT = WvT + WD;
  u16* Qb  = WoT + WD;        // phi(xWq)              [MS x D]
  u16* Kb  = Qb  + ND;
  u16* Vb  = Kb  + ND;
  u16* Ob  = Vb  + ND;        // attention output      [MS x D]

  // 1) convert x
  cvt_x_kernel<<<dim3((unsigned)((ND + 255) / 256)), dim3(256), 0, stream>>>(x, xb, ND);

  // 2) convert + transpose weights
  dim3 tg(Dm / 32, Dm / 32), tb(32, 8);
  cvt_wT_kernel<<<tg, tb, 0, stream>>>(Wq, WqT, Dm);
  cvt_wT_kernel<<<tg, tb, 0, stream>>>(Wk, WkT, Dm);
  cvt_wT_kernel<<<tg, tb, 0, stream>>>(Wv, WvT, Dm);
  cvt_wT_kernel<<<tg, tb, 0, stream>>>(Wo, WoT, Dm);

  // 3) projections (phi fused for Q,K)
  dim3 gg((unsigned)(MS / GT_M), (unsigned)(Dm / GT_N));
  gemm_bf16_kernel<<<gg, 256, 0, stream>>>(xb, WqT, Qb, nullptr, (int)MS, Dm, Dm, 1);
  gemm_bf16_kernel<<<gg, 256, 0, stream>>>(xb, WkT, Kb, nullptr, (int)MS, Dm, Dm, 1);
  gemm_bf16_kernel<<<gg, 256, 0, stream>>>(xb, WvT, Vb, nullptr, (int)MS, Dm, Dm, 0);

  // 4) chunked recurrent linear attention (one WG per batch*head)
  attn_chunk_kernel<<<dim3(Bb * H), dim3(128), 0, stream>>>(Qb, Kb, Vb, Ob, S, Dm, H);

  // 5) output projection -> f32
  gemm_bf16_kernel<<<gg, 256, 0, stream>>>(Ob, WoT, nullptr, out, (int)MS, Dm, Dm, 2);
}
